// ProteinGraphConv_84241488544324
// MI455X (gfx1250) — matrix operand order repaired and checked by reading, not compiled, not run on hardware
//
#include <hip/hip_runtime.h>
#include <hip/hip_bf16.h>
#include <math.h>

// ---------- CDNA5 WMMA types ----------
typedef __attribute__((ext_vector_type(16))) __bf16 v16bf;
typedef __attribute__((ext_vector_type(8)))  float  v8f;

#define ND 128
#define HD 128
#define ED 32

__device__ __forceinline__ float gelu_exact(float x) {
    // torch nn.GELU default (exact): 0.5*x*(1+erf(x/sqrt(2)))
    return 0.5f * x * (1.0f + erff(x * 0.70710678118654752f));
}

// ------------------------------------------------------------------
// Pack a K x Nfull f32 weight matrix into bf16 WMMA B-fragments.
// Fragment layout (16x16x32 bf16 B-matrix, wave32):
//   entry j of lane l in slice s, n-tile t  ->  W[(s*32 + j + (j>=8 ? 8 : 0) + 8*(l>>4)) * Nfull + (l&15) + 16*t]
// Stored so each lane reads one contiguous 32B v16bf.
// ------------------------------------------------------------------
__global__ void pack_b_kernel(const float* __restrict__ W, __bf16* __restrict__ out,
                              int K, int Nfull) {
    const int ns = K >> 5;                 // K/32 slices
    const int total = (Nfull >> 4) * ns * 32 * 16;
    int idx = blockIdx.x * 256 + threadIdx.x;
    if (idx >= total) return;
    int j = idx & 15;
    int l = (idx >> 4) & 31;
    int rest = idx >> 9;
    int s = rest % ns;
    int t = rest / ns;
    int k = s * 32 + j + ((j >= 8) ? 8 : 0) + 8 * (l >> 4);
    int n = (l & 15) + 16 * t;
    out[idx] = (__bf16)W[(long)k * Nfull + n];
}

__global__ void zero_kernel(float* __restrict__ p, long n) {
    long i = (long)blockIdx.x * 256 + threadIdx.x;
    if (i < n) p[i] = 0.0f;
}

// ------------------------------------------------------------------
// Edge message kernel: per wave, 32 edges (two 16-row A tiles, register
// blocked in M so each loaded B fragment feeds TWO WMMAs).
//   msg = GELU(GELU([nf[src]|nf[dst]|ef] @ W1 + b1) @ W2 + b2)
//   atomicAdd into agg[dst].
// K1 = 288 (9 slices), K2 = 128 (4 slices), N = 128 (8 n-tiles).
// ------------------------------------------------------------------
__global__ void __launch_bounds__(256)
edge_msg_kernel(const float* __restrict__ nf,
                const int*   __restrict__ src,
                const int*   __restrict__ dst,
                const float* __restrict__ ef,
                const float* __restrict__ b1,
                const float* __restrict__ b2,
                const v16bf* __restrict__ W1b,
                const v16bf* __restrict__ W2b,
                float* __restrict__ agg,
                int E) {
    __shared__ __bf16 lds_h[8][32 * HD];   // per-wave hidden tile (32 x 128 bf16), 64 KB total

    const int lane = threadIdx.x & 31;
    const int wave = threadIdx.x >> 5;
    const int m = lane & 15;               // A-matrix row handled by this lane
    const int g = lane >> 4;               // lane half
    const long ebase = ((long)blockIdx.x * 8 + wave) * 32;

    // two edges gathered per lane (one per M-tile); clamp keeps EXEC all-ones
    long eidx[2];
    int  sidx[2], didx[2];
#pragma unroll
    for (int h = 0; h < 2; ++h) {
        long e = ebase + 16 * h + m;
        if (e >= E) e = E - 1;
        eidx[h] = e;
        sidx[h] = src[e];
        didx[h] = dst[e];
    }

    // ---- Load 2 x 9 A-fragments (16x32 bf16 each) from the virtual concat input.
    // A-layout: entry j of lane -> k_in_slice = j + (j>=8 ? 8 : 0) + 8*g.
    v16bf a[2][9];
#pragma unroll
    for (int h = 0; h < 2; ++h) {
        const long e = eidx[h];
        const long sb = (long)sidx[h] * ND;
        const long db = (long)didx[h] * ND;
#pragma unroll
        for (int s = 0; s < 9; ++s) {
#pragma unroll
            for (int j = 0; j < 16; j += 2) {
                const int C = s * 32 + j + ((j >= 8) ? 8 : 0);  // compile-time after unroll
                float x0, x1;
                if (C < 128) {            // source node features
                    int k = C + 8 * g;
                    x0 = nf[sb + k];
                    x1 = nf[sb + k + 1];
                } else if (C < 256) {     // destination node features
                    int k = C - 128 + 8 * g;
                    x0 = nf[db + k];
                    x1 = nf[db + k + 1];
                } else {                  // edge features
                    int k = C - 256 + 8 * g;
                    x0 = ef[e * ED + k];
                    x1 = ef[e * ED + k + 1];
                }
                a[h][s][j]     = (__bf16)x0;
                a[h][s][j + 1] = (__bf16)x1;
            }
        }
    }

    // ---- GEMM1 (+b1, GELU) -> LDS hidden (bf16, row-major 32x128)
#pragma unroll
    for (int t = 0; t < 8; ++t) {
        const float bias = b1[m + 16 * t];
        v8f c0, c1;
#pragma unroll
        for (int v = 0; v < 8; ++v) { c0[v] = bias; c1[v] = bias; }
#pragma unroll
        for (int s = 0; s < 9; ++s) {
            v16bf b = W1b[(t * 9 + s) * 32 + lane];       // one B load ...
            c0 = __builtin_amdgcn_wmma_f32_16x16x32_bf16(false, a[0][s], false, b,
                                                         (short)0, c0, false, false);
            c1 = __builtin_amdgcn_wmma_f32_16x16x32_bf16(false, a[1][s], false, b,
                                                         (short)0, c1, false, false);  // ... two WMMAs
        }
#pragma unroll
        for (int v = 0; v < 8; ++v) {
            const int M = v + 8 * g;       // C layout: row = v + 8*half, col = m + 16*t
            lds_h[wave][M * HD + m + 16 * t]        = (__bf16)gelu_exact(c0[v]);
            lds_h[wave][(16 + M) * HD + m + 16 * t] = (__bf16)gelu_exact(c1[v]);
        }
    }

    // ---- Re-load hidden as A-fragments for GEMM2 (K=128 -> 4 slices)
    v16bf a2[2][4];
#pragma unroll
    for (int h = 0; h < 2; ++h) {
#pragma unroll
        for (int s = 0; s < 4; ++s) {
#pragma unroll
            for (int j = 0; j < 16; ++j) {
                const int k = s * 32 + j + ((j >= 8) ? 8 : 0) + 8 * g;
                a2[h][s][j] = lds_h[wave][(16 * h + m) * HD + k];
            }
        }
    }

    // dst index per output row (row = v + 8*g within each half-tile)
    int drow[2][8];
#pragma unroll
    for (int h = 0; h < 2; ++h) {
#pragma unroll
        for (int v = 0; v < 8; ++v) {
            const long er = ebase + 16 * h + v + 8 * g;
            drow[h][v] = (er < E) ? dst[er] : -1;
        }
    }

    // ---- GEMM2 (+b2, GELU) -> scatter-add into agg
#pragma unroll
    for (int t = 0; t < 8; ++t) {
        const float bias = b2[m + 16 * t];
        v8f c0, c1;
#pragma unroll
        for (int v = 0; v < 8; ++v) { c0[v] = bias; c1[v] = bias; }
#pragma unroll
        for (int s = 0; s < 4; ++s) {
            v16bf b = W2b[(t * 4 + s) * 32 + lane];
            c0 = __builtin_amdgcn_wmma_f32_16x16x32_bf16(false, a2[0][s], false, b,
                                                         (short)0, c0, false, false);
            c1 = __builtin_amdgcn_wmma_f32_16x16x32_bf16(false, a2[1][s], false, b,
                                                         (short)0, c1, false, false);
        }
#pragma unroll
        for (int v = 0; v < 8; ++v) {
            if (drow[0][v] >= 0)
                atomicAdd(&agg[(long)drow[0][v] * HD + m + 16 * t], gelu_exact(c0[v]));
            if (drow[1][v] >= 0)
                atomicAdd(&agg[(long)drow[1][v] * HD + m + 16 * t], gelu_exact(c1[v]));
        }
    }
}

// ------------------------------------------------------------------
// Node update kernel: per wave, 16 nodes.
//   upd = nf + GELU([nf|agg] @ U1 + c1) @ U2 + c2 ; LayerNorm ; mask
// K1 = 256 (8 slices), K2 = 128 (4 slices), N = 128 (8 n-tiles).
// ------------------------------------------------------------------
__global__ void __launch_bounds__(128)
node_update_kernel(const float* __restrict__ nf,
                   const float* __restrict__ agg,
                   const float* __restrict__ c1,
                   const float* __restrict__ c2,
                   const v16bf* __restrict__ U1b,
                   const v16bf* __restrict__ U2b,
                   const float* __restrict__ gamma,
                   const float* __restrict__ beta,
                   const unsigned char* __restrict__ mask,
                   float* __restrict__ out,
                   int Nn) {
    __shared__ __bf16 lds_h[4][16 * HD];   // hidden bf16, 16 KB
    __shared__ float  lds_u[4][16 * ND];   // updated f32, 32 KB

    const int lane = threadIdx.x & 31;
    const int wave = threadIdx.x >> 5;
    const int m = lane & 15;
    const int g = lane >> 4;
    const long nbase = ((long)blockIdx.x * 4 + wave) * 16;

    long node = nbase + m;
    if (node >= Nn) node = Nn - 1;         // clamp: EXEC stays all-ones

    // ---- A-fragments from [nf | agg] (K = 256)
    v16bf a[8];
#pragma unroll
    for (int s = 0; s < 8; ++s) {
#pragma unroll
        for (int j = 0; j < 16; j += 2) {
            const int C = s * 32 + j + ((j >= 8) ? 8 : 0);
            float x0, x1;
            if (C < 128) {
                int k = C + 8 * g;
                x0 = nf[node * ND + k];
                x1 = nf[node * ND + k + 1];
            } else {
                int k = C - 128 + 8 * g;
                x0 = agg[node * HD + k];
                x1 = agg[node * HD + k + 1];
            }
            a[s][j]     = (__bf16)x0;
            a[s][j + 1] = (__bf16)x1;
        }
    }

    // ---- U1 (+c1, GELU) -> LDS hidden
#pragma unroll
    for (int t = 0; t < 8; ++t) {
        const float bias = c1[m + 16 * t];
        v8f c;
#pragma unroll
        for (int v = 0; v < 8; ++v) c[v] = bias;
#pragma unroll
        for (int s = 0; s < 8; ++s) {
            v16bf b = U1b[(t * 8 + s) * 32 + lane];
            c = __builtin_amdgcn_wmma_f32_16x16x32_bf16(false, a[s], false, b,
                                                        (short)0, c, false, false);
        }
#pragma unroll
        for (int v = 0; v < 8; ++v) {
            const int M = v + 8 * g;
            lds_h[wave][M * HD + m + 16 * t] = (__bf16)gelu_exact(c[v]);
        }
    }

    // ---- hidden -> A-fragments for U2
    v16bf a2[4];
#pragma unroll
    for (int s = 0; s < 4; ++s) {
#pragma unroll
        for (int j = 0; j < 16; ++j) {
            const int k = s * 32 + j + ((j >= 8) ? 8 : 0) + 8 * g;
            a2[s][j] = lds_h[wave][m * HD + k];
        }
    }

    // ---- U2 (+c2) + residual -> LDS f32
#pragma unroll
    for (int t = 0; t < 8; ++t) {
        const float bias = c2[m + 16 * t];
        v8f c;
#pragma unroll
        for (int v = 0; v < 8; ++v) c[v] = bias;
#pragma unroll
        for (int s = 0; s < 4; ++s) {
            v16bf b = U2b[(t * 4 + s) * 32 + lane];
            c = __builtin_amdgcn_wmma_f32_16x16x32_bf16(false, a2[s], false, b,
                                                        (short)0, c, false, false);
        }
#pragma unroll
        for (int v = 0; v < 8; ++v) {
            const int M = v + 8 * g;
            const long nd = nbase + M;
            const float r = (nd < Nn) ? nf[nd * ND + m + 16 * t] : 0.0f;
            lds_u[wave][M * ND + m + 16 * t] = c[v] + r;
        }
    }

    // ---- LayerNorm + mask + store (lanes 0..15, one row each; WMMAs are done)
    if (g == 0) {
        const long nd = nbase + m;
        if (nd < Nn) {
            float mu = 0.0f;
#pragma unroll 4
            for (int k = 0; k < ND; ++k) mu += lds_u[wave][m * ND + k];
            mu *= (1.0f / ND);
            float var = 0.0f;
#pragma unroll 4
            for (int k = 0; k < ND; ++k) {
                const float d = lds_u[wave][m * ND + k] - mu;
                var += d * d;
            }
            var *= (1.0f / ND);
            const float inv = rsqrtf(var + 1e-5f);
            const float mk = mask[nd] ? 1.0f : 0.0f;
#pragma unroll 4
            for (int k = 0; k < ND; ++k) {
                const float x = lds_u[wave][m * ND + k];
                out[nd * ND + k] = ((x - mu) * inv * gamma[k] + beta[k]) * mk;
            }
        }
    }
}

// ------------------------------------------------------------------
extern "C" void kernel_launch(void* const* d_in, const int* in_sizes, int n_in,
                              void* d_out, int out_size, void* d_ws, size_t ws_size,
                              hipStream_t stream) {
    const float* nf   = (const float*)d_in[0];
    const int*   ei   = (const int*)d_in[1];
    const float* ef   = (const float*)d_in[2];
    const unsigned char* mask = (const unsigned char*)d_in[3];
    const float* W1 = (const float*)d_in[4];
    const float* b1 = (const float*)d_in[5];
    const float* W2 = (const float*)d_in[6];
    const float* b2 = (const float*)d_in[7];
    const float* U1 = (const float*)d_in[8];
    const float* c1 = (const float*)d_in[9];
    const float* U2 = (const float*)d_in[10];
    const float* c2 = (const float*)d_in[11];
    const float* gamma = (const float*)d_in[12];
    const float* beta  = (const float*)d_in[13];

    const int Nn = in_sizes[0] / ND;   // 50000
    const int E  = in_sizes[1] / 2;    // 800000
    const int* src = ei;
    const int* dst = ei + E;

    // Workspace layout (all 32B aligned):
    float*  agg = (float*)d_ws;                         // Nn*HD f32
    __bf16* W1b = (__bf16*)(agg + (size_t)Nn * HD);     // 8*9*32*16
    __bf16* W2b = W1b + 8 * 9 * 32 * 16;                // 8*4*32*16
    __bf16* U1b = W2b + 8 * 4 * 32 * 16;                // 8*8*32*16
    __bf16* U2b = U1b + 8 * 8 * 32 * 16;                // 8*4*32*16

    const long aggN = (long)Nn * HD;
    zero_kernel<<<(int)((aggN + 255) / 256), 256, 0, stream>>>(agg, aggN);
    pack_b_kernel<<<(288 * 128 + 255) / 256, 256, 0, stream>>>(W1, W1b, 288, 128);
    pack_b_kernel<<<(128 * 128 + 255) / 256, 256, 0, stream>>>(W2, W2b, 128, 128);
    pack_b_kernel<<<(256 * 128 + 255) / 256, 256, 0, stream>>>(U1, U1b, 256, 128);
    pack_b_kernel<<<(128 * 128 + 255) / 256, 256, 0, stream>>>(U2, U2b, 128, 128);

    const int etiles  = (E + 31) / 32;               // 32 edges per wave
    const int eblocks = (etiles + 7) / 8;            // 8 waves per block
    edge_msg_kernel<<<eblocks, 256, 0, stream>>>(nf, src, dst, ef, b1, b2,
                                                 (const v16bf*)W1b, (const v16bf*)W2b,
                                                 agg, E);

    const int nwaves  = (Nn + 15) / 16;
    const int nblocks = (nwaves + 3) / 4;
    node_update_kernel<<<nblocks, 128, 0, stream>>>(nf, agg, c1, c2,
                                                    (const v16bf*)U1b, (const v16bf*)U2b,
                                                    gamma, beta, mask,
                                                    (float*)d_out, Nn);
}
